// ConvLinearLayer_39908836114670
// MI455X (gfx1250) — matrix operand order repaired and checked by reading, not compiled, not run on hardware
//
#include <hip/hip_runtime.h>
#include <hip/hip_bf16.h>
#include <math.h>
#include <stdint.h>

// ---------------------------------------------------------------------------
// Types / helpers
// ---------------------------------------------------------------------------
typedef __bf16 bf16_t;
typedef bf16_t v16bf __attribute__((ext_vector_type(16)));
typedef float  v8f   __attribute__((ext_vector_type(8)));
typedef unsigned short u16;

__device__ __forceinline__ u16 f2bf_bits(float f) {
  unsigned u = __float_as_uint(f);
  u += 0x7FFFu + ((u >> 16) & 1u);          // round-to-nearest-even
  return (u16)(u >> 16);
}
__device__ __forceinline__ float bf2f(u16 h) {
  return __uint_as_float(((unsigned)h) << 16);
}

enum { MODE_MLL = 0, MODE_BIAS_RELU = 1, MODE_BIAS = 2 };

// ---------------------------------------------------------------------------
// WMMA GEMM: out[M x N] = epilogue(A[M x K] @ W[N x K]^T)
//   block = 256 threads (8 wave32), tile = 128 rows x 64 cols
//   wave w -> rows [w*16, w*16+16), 4 column sub-tiles of 16
//   MODE_MLL: out = silu(A@Wb^T) + A@Ws^T       (Ws pre-scaled by 1/K_SPLINE)
//   MODE_BIAS[_RELU]: out = [relu](A@Wb^T + bias)
//
// A tile is DMA'd global->LDS with GLOBAL_LOAD_ASYNC_TO_LDS_B128 (ASYNCcnt),
// double-buffered so the copy for step i+1 overlaps the WMMAs of step i.
// W fragments read directly from global (weights are L2-resident).
// K % 32 == 0, M % 128 == 0, N % 64 == 0 (true for every layer here).
// ---------------------------------------------------------------------------
template<int MODE>
__global__ __launch_bounds__(256, 2)
void gemm_wmma_bf16(const u16* __restrict__ A,
                    const u16* __restrict__ Wb,
                    const u16* __restrict__ Ws,
                    const float* __restrict__ bias,
                    u16* __restrict__ out,
                    int K, int OS, int coff)
{
  __shared__ u16 As[2][128 * 32];

  const int tid   = threadIdx.x;
  const int wv    = tid >> 5;
  const int lane  = tid & 31;
  const int khalf = lane >> 4;       // 0 or 1 (K-half selector in this code's contiguous k order)
  const int l16   = lane & 15;
  const long m0   = (long)blockIdx.x * 128;
  const int  n0   = blockIdx.y * 64;

  v8f accB[4], accS[4];
  #pragma unroll
  for (int t = 0; t < 4; ++t)
    #pragma unroll
    for (int j = 0; j < 8; ++j) { accB[t][j] = 0.f; accS[t][j] = 0.f; }

  // Per-thread async-copy slice: 32 bytes of one A-tile row.
  const int r = tid >> 1;            // 0..127
  const int h = (tid & 1) << 4;      // 0 or 16 (bf16 elements)
  const u16* gsrc = A + (m0 + r) * (long)K + h;

  auto issue_tile = [&](int k0, int buf) {
    // LDS byte offset = low 32 bits of the generic pointer (LDS aperture).
    const unsigned lds = (unsigned)(uintptr_t)(&As[buf][r * 32 + h]);
    const unsigned long long g = (unsigned long long)(uintptr_t)(gsrc + k0);
    asm volatile(
        "global_load_async_to_lds_b128 %0, %1, off\n\t"
        "global_load_async_to_lds_b128 %0, %1, off offset:16"
        :: "v"(lds), "v"(g) : "memory");
  };

  const int lrow  = wv * 16 + l16;
  const int nIter = K >> 5;

  issue_tile(0, 0);

  for (int it = 0; it < nIter; ++it) {
    const int buf = it & 1;
    const int k0  = it << 5;

    // Own async done -> barrier -> every wave's slice of As[buf] is visible.
    asm volatile("s_wait_asynccnt 0x0" ::: "memory");
    __syncthreads();

    // Prefetch next tile into the other buffer; everyone has finished
    // reading it (it was consumed in iteration it-1, before this barrier).
    if (it + 1 < nIter) issue_tile(k0 + 32, buf ^ 1);

    // A fragment (16x32 bf16): lane<16 -> K 0-15, lane>=16 -> K 16-31
    union { v16bf v; uint4 u[2]; } af;
    af.u[0] = *(const uint4*)(&As[buf][lrow * 32 + khalf * 16]);
    af.u[1] = *(const uint4*)(&As[buf][lrow * 32 + khalf * 16 + 8]);

    #pragma unroll
    for (int t = 0; t < 4; ++t) {
      // B fragment (32x16): lane holds 16 contiguous K of weight-row n
      const long nrow = (long)(n0 + t * 16 + l16);
      v16bf bb = *(const v16bf*)(Wb + nrow * K + k0 + khalf * 16);
      accB[t] = __builtin_amdgcn_wmma_f32_16x16x32_bf16(
          false, af.v, false, bb, (short)0, accB[t], false, false);
      if (MODE == MODE_MLL) {
        v16bf bs = *(const v16bf*)(Ws + nrow * K + k0 + khalf * 16);
        accS[t] = __builtin_amdgcn_wmma_f32_16x16x32_bf16(
            false, af.v, false, bs, (short)0, accS[t], false, false);
      }
    }
  }

  // C/D layout: element j of v8f @ lane l -> row j + (l>>4)*8, col l&15
  const long rbase = m0 + wv * 16 + khalf * 8;
  #pragma unroll
  for (int t = 0; t < 4; ++t) {
    const int ncol = n0 + t * 16 + l16;
    const float bv = (MODE == MODE_MLL) ? 0.f : bias[ncol];
    #pragma unroll
    for (int j = 0; j < 8; ++j) {
      const float b = accB[t][j];
      float o;
      if (MODE == MODE_MLL)            o = b / (1.f + __expf(-b)) + accS[t][j];
      else if (MODE == MODE_BIAS_RELU) o = fmaxf(b + bv, 0.f);
      else                             o = b + bv;
      out[(rbase + j) * (long)OS + coff + ncol] = f2bf_bits(o);
    }
  }
}

// ---------------------------------------------------------------------------
// Depthwise 3x3 conv ('SAME', zero pad) over [B,N,C] channel-last, H=W=64.
// grid.x = spatial index (B*N), block = C threads (coalesced over c).
// ---------------------------------------------------------------------------
__global__ void dwconv3x3(const u16* __restrict__ xin, const float* __restrict__ wgt,
                          const float* __restrict__ bias, float* __restrict__ y, int C)
{
  const int s = blockIdx.x;
  const int c = threadIdx.x;
  const int b = s >> 12;
  const int h = (s >> 6) & 63;
  const int w = s & 63;
  float acc = bias[c];
  const float* wc = wgt + c * 9;
  #pragma unroll
  for (int dh = -1; dh <= 1; ++dh) {
    const int hh = h + dh;
    if ((unsigned)hh >= 64u) continue;
    #pragma unroll
    for (int dw = -1; dw <= 1; ++dw) {
      const int ww = w + dw;
      if ((unsigned)ww >= 64u) continue;
      const long idx = (long)((b << 12) + (hh << 6) + ww) * C + c;
      acc += bf2f(xin[idx]) * wc[(dh + 1) * 3 + (dw + 1)];
    }
  }
  y[(long)s * C + c] = acc;
}

// Train-mode BN stats: per-channel sum & sum-of-squares over 65536 positions.
__global__ void bn_stats(const float* __restrict__ y, float* __restrict__ stats, int C)
{
  const int c = blockIdx.x;
  __shared__ float s1[256], s2[256];
  float a = 0.f, b = 0.f;
  const long base = (long)blockIdx.y * 1024 + threadIdx.x * 4;
  #pragma unroll
  for (int j = 0; j < 4; ++j) {
    const float v = y[(base + j) * C + c];
    a += v; b += v * v;
  }
  s1[threadIdx.x] = a; s2[threadIdx.x] = b;
  __syncthreads();
  for (int off = 128; off > 0; off >>= 1) {
    if (threadIdx.x < off) {
      s1[threadIdx.x] += s1[threadIdx.x + off];
      s2[threadIdx.x] += s2[threadIdx.x + off];
    }
    __syncthreads();
  }
  if (threadIdx.x == 0) {
    atomicAdd(&stats[c], s1[0]);
    atomicAdd(&stats[C + c], s2[0]);
  }
}

__global__ void bn_finalize(float* __restrict__ stats, const float* __restrict__ gamma,
                            const float* __restrict__ beta, int C)
{
  const int c = blockIdx.x * blockDim.x + threadIdx.x;
  if (c >= C) return;
  const float invN = 1.f / 65536.f;
  const float mean = stats[c] * invN;
  const float var  = stats[C + c] * invN - mean * mean;   // biased, as torch BN train
  const float inv  = rsqrtf(var + 1e-5f);
  const float a = gamma[c] * inv;
  stats[2 * C + c] = a;
  stats[3 * C + c] = beta[c] - a * mean;
}

// Apply BN + ReLU, write bf16 into strided destination (fuses the concat).
__global__ void bn_apply_relu_bf16(const float* __restrict__ y, const float* __restrict__ stats,
                                   u16* __restrict__ dst, int C, int OS, int coff)
{
  const int s = blockIdx.x;
  const int c = threadIdx.x;
  const float v = fmaxf(stats[2 * C + c] * y[(long)s * C + c] + stats[3 * C + c], 0.f);
  dst[(long)s * OS + coff + c] = f2bf_bits(v);
}

// Final BN + ReLU + scaled residual, f32 output.
__global__ void bn_apply_final(const float* __restrict__ y, const float* __restrict__ stats,
                               const float* __restrict__ xin, const float* __restrict__ res_scale,
                               float* __restrict__ out)
{
  const int C = 512;
  const int s = blockIdx.x;
  const int c = threadIdx.x;
  const long i = (long)s * C + c;
  const float v = fmaxf(stats[2 * C + c] * y[i] + stats[3 * C + c], 0.f);
  out[i] = v + res_scale[0] * xin[i];
}

// ---------------------------------------------------------------------------
// Weight prep: MLL spline collapse (sum over K_SPLINE, /10) + bf16 convert;
// channel_scale folded into fus_w1 columns.
// ---------------------------------------------------------------------------
__global__ void prep_mll_w(const float* __restrict__ bw, const float* __restrict__ sw,
                           u16* __restrict__ wb, u16* __restrict__ ws, int total)
{
  const int i = blockIdx.x * blockDim.x + threadIdx.x;
  if (i >= total) return;
  wb[i] = f2bf_bits(bw[i]);
  float s = 0.f;
  #pragma unroll
  for (int t = 0; t < 10; ++t) s += sw[i * 10 + t];
  ws[i] = f2bf_bits(s * 0.1f);
}

__global__ void prep_scaled_w(const float* __restrict__ w, const float* __restrict__ cs,
                              u16* __restrict__ o, int total, int K)
{
  const int i = blockIdx.x * blockDim.x + threadIdx.x;
  if (i >= total) return;
  const float scale = cs ? cs[i % K] : 1.f;
  o[i] = f2bf_bits(w[i] * scale);
}

__global__ void f32_to_bf16(const float* __restrict__ x, u16* __restrict__ o, long n)
{
  const long i = (long)blockIdx.x * blockDim.x + threadIdx.x;
  if (i < n) o[i] = f2bf_bits(x[i]);
}

__global__ void zero_f32(float* __restrict__ p, int n)
{
  for (int i = blockIdx.x * blockDim.x + threadIdx.x; i < n; i += gridDim.x * blockDim.x)
    p[i] = 0.f;
}

// ---------------------------------------------------------------------------
// Host orchestration
// ---------------------------------------------------------------------------
extern "C" void kernel_launch(void* const* d_in, const int* in_sizes, int n_in,
                              void* d_out, int out_size, void* d_ws, size_t ws_size,
                              hipStream_t stream)
{
  (void)in_sizes; (void)n_in; (void)out_size; (void)ws_size;

  const float* x        = (const float*)d_in[0];
  const float* fc1l_bw  = (const float*)d_in[3];
  const float* fc1l_sw  = (const float*)d_in[4];
  const float* fc2l_bw  = (const float*)d_in[5];
  const float* fc2l_sw  = (const float*)d_in[6];
  const float* fc1f_bw  = (const float*)d_in[7];
  const float* fc1f_sw  = (const float*)d_in[8];
  const float* fc3_bw   = (const float*)d_in[9];
  const float* fc3_sw   = (const float*)d_in[10];
  const float* fus_w1   = (const float*)d_in[11];
  const float* fus_b1   = (const float*)d_in[12];
  const float* fus_w2   = (const float*)d_in[13];
  const float* fus_b2   = (const float*)d_in[14];
  const float* cs       = (const float*)d_in[15];
  const float* dw1_w    = (const float*)d_in[16];
  const float* dw1_b    = (const float*)d_in[17];
  const float* dw1_g    = (const float*)d_in[18];
  const float* dw1_beta = (const float*)d_in[19];
  const float* dw2_w    = (const float*)d_in[20];
  const float* dw2_b    = (const float*)d_in[21];
  const float* dw2_g    = (const float*)d_in[22];
  const float* dw2_beta = (const float*)d_in[23];
  const float* dw3_w    = (const float*)d_in[24];
  const float* dw3_b    = (const float*)d_in[25];
  const float* dw3_g    = (const float*)d_in[26];
  const float* dw3_beta = (const float*)d_in[27];
  const float* res_scale= (const float*)d_in[28];
  float* out = (float*)d_out;

  const long M = 65536;                     // B*N = 16*4096
  char* ws = (char*)d_ws;
  size_t off = 0;
  auto alloc = [&](size_t bytes) -> void* {
    void* p = ws + off;
    off = (off + bytes + 255) & ~(size_t)255;
    return p;
  };

  u16* wb_f1l = (u16*)alloc(128 * 512 * 2);
  u16* ws_f1l = (u16*)alloc(128 * 512 * 2);
  u16* wb_f1f = (u16*)alloc(256 * 512 * 2);
  u16* ws_f1f = (u16*)alloc(256 * 512 * 2);
  u16* wb_f2  = (u16*)alloc(128 * 128 * 2);
  u16* ws_f2  = (u16*)alloc(128 * 128 * 2);
  u16* wb_f3  = (u16*)alloc(512 * 384 * 2);
  u16* ws_f3  = (u16*)alloc(512 * 384 * 2);
  u16* w_u1   = (u16*)alloc(192 * 384 * 2);
  u16* w_u2   = (u16*)alloc(384 * 192 * 2);
  float* stats= (float*)alloc(4 * 512 * 4);
  u16* xbf  = (u16*)alloc((size_t)M * 512 * 2);
  u16* a1l  = (u16*)alloc((size_t)M * 128 * 2);
  u16* a1f  = (u16*)alloc((size_t)M * 256 * 2);
  u16* c1o  = (u16*)alloc((size_t)M * 128 * 2);
  u16* hcat = (u16*)alloc((size_t)M * 384 * 2);
  u16* fmid = (u16*)alloc((size_t)M * 192 * 2);
  u16* hf   = (u16*)alloc((size_t)M * 384 * 2);
  u16* f3o  = (u16*)alloc((size_t)M * 512 * 2);
  float* cvy= (float*)alloc((size_t)M * 512 * 4);

  // ---- weight prep + activation convert -------------------------------
  prep_mll_w<<<(128 * 512 + 255) / 256, 256, 0, stream>>>(fc1l_bw, fc1l_sw, wb_f1l, ws_f1l, 128 * 512);
  prep_mll_w<<<(256 * 512 + 255) / 256, 256, 0, stream>>>(fc1f_bw, fc1f_sw, wb_f1f, ws_f1f, 256 * 512);
  prep_mll_w<<<(128 * 128 + 255) / 256, 256, 0, stream>>>(fc2l_bw, fc2l_sw, wb_f2, ws_f2, 128 * 128);
  prep_mll_w<<<(512 * 384 + 255) / 256, 256, 0, stream>>>(fc3_bw, fc3_sw, wb_f3, ws_f3, 512 * 384);
  prep_scaled_w<<<(192 * 384 + 255) / 256, 256, 0, stream>>>(fus_w1, cs, w_u1, 192 * 384, 384);
  prep_scaled_w<<<(384 * 192 + 255) / 256, 256, 0, stream>>>(fus_w2, nullptr, w_u2, 384 * 192, 192);
  f32_to_bf16<<<(unsigned)((M * 512 + 255) / 256), 256, 0, stream>>>(x, xbf, M * 512);

  // ---- fc1_low / fc1_full (MLL) ---------------------------------------
  gemm_wmma_bf16<MODE_MLL><<<dim3(512, 2), 256, 0, stream>>>(xbf, wb_f1l, ws_f1l, nullptr, a1l, 512, 128, 0);
  gemm_wmma_bf16<MODE_MLL><<<dim3(512, 4), 256, 0, stream>>>(xbf, wb_f1f, ws_f1f, nullptr, a1f, 512, 256, 0);

  // ---- dwconv1 + BN + ReLU (C=128) ------------------------------------
  zero_f32<<<2, 256, 0, stream>>>(stats, 2 * 128);
  dwconv3x3<<<65536, 128, 0, stream>>>(a1l, dw1_w, dw1_b, cvy, 128);
  bn_stats<<<dim3(128, 64), 256, 0, stream>>>(cvy, stats, 128);
  bn_finalize<<<1, 128, 0, stream>>>(stats, dw1_g, dw1_beta, 128);
  bn_apply_relu_bf16<<<65536, 128, 0, stream>>>(cvy, stats, c1o, 128, 128, 0);

  // ---- fc2_low (MLL) -> hcat columns [0,128) --------------------------
  gemm_wmma_bf16<MODE_MLL><<<dim3(512, 2), 256, 0, stream>>>(c1o, wb_f2, ws_f2, nullptr, hcat, 128, 384, 0);

  // ---- dwconv2 + BN + ReLU (C=256) -> hcat columns [128,384) ----------
  zero_f32<<<2, 256, 0, stream>>>(stats, 2 * 256);
  dwconv3x3<<<65536, 256, 0, stream>>>(a1f, dw2_w, dw2_b, cvy, 256);
  bn_stats<<<dim3(256, 64), 256, 0, stream>>>(cvy, stats, 256);
  bn_finalize<<<1, 256, 0, stream>>>(stats, dw2_g, dw2_beta, 256);
  bn_apply_relu_bf16<<<65536, 256, 0, stream>>>(cvy, stats, hcat, 256, 384, 128);

  // ---- fusion MLP (channel_scale pre-folded into w_u1) ----------------
  gemm_wmma_bf16<MODE_BIAS_RELU><<<dim3(512, 3), 256, 0, stream>>>(hcat, w_u1, nullptr, fus_b1, fmid, 384, 192, 0);
  gemm_wmma_bf16<MODE_BIAS><<<dim3(512, 6), 256, 0, stream>>>(fmid, w_u2, nullptr, fus_b2, hf, 192, 384, 0);

  // ---- fc3 (MLL) ------------------------------------------------------
  gemm_wmma_bf16<MODE_MLL><<<dim3(512, 8), 256, 0, stream>>>(hf, wb_f3, ws_f3, nullptr, f3o, 384, 512, 0);

  // ---- dwconv3 + BN + ReLU + residual (C=512) -> f32 out --------------
  zero_f32<<<4, 256, 0, stream>>>(stats, 2 * 512);
  dwconv3x3<<<65536, 512, 0, stream>>>(f3o, dw3_w, dw3_b, cvy, 512);
  bn_stats<<<dim3(512, 64), 256, 0, stream>>>(cvy, stats, 512);
  bn_finalize<<<1, 512, 0, stream>>>(stats, dw3_g, dw3_beta, 512);
  bn_apply_final<<<65536, 512, 0, stream>>>(cvy, stats, x, res_scale, out);
}
